// RCP_O4_Conv2d_86363202388467
// MI455X (gfx1250) — matrix operand order repaired and checked by reading, not compiled, not run on hardware
//
#include <hip/hip_runtime.h>

typedef __attribute__((ext_vector_type(16))) _Float16 v16h;
typedef __attribute__((ext_vector_type(8)))  _Float16 v8h;
typedef __attribute__((ext_vector_type(8)))  float    v8f;

// ---------------- tile geometry ----------------
#define TILE_W   128     // output pixels per block (along W)
#define TILE_H   8       // output rows per block (one per wave)
#define HALO     10      // staged input rows  (TILE_H + 2)
#define PXS      130     // staged input pixels per row (TILE_W + 2)
#define PXSTRIDE 40      // halves per pixel in LDS (32 ch + 8 pad -> 80B, conflict-free b128)
#define WSTRIDE  40      // halves per oc row in weight LDS (same padding trick)

#define XS_HALVES (HALO * PXS * PXSTRIDE)   // 52000
#define WL_HALVES (9 * 32 * WSTRIDE)        // 11520

// ---------------------------------------------------------------
// Kernel 1: CP weight reconstruction  W[tap][oc][ic] (f16, dense)
//   W[o,c,kh,kw] = sum_r k0[i,a,r] k1[j,b,r] k2[k,cc,r] k3[kh,kw,r]
//   o = i*16 + j*4 + k   (i<2, j<4, k<4)
//   c = a*16 + b*4 + cc  (a<2, b<4, cc<4)
// ---------------------------------------------------------------
__global__ __launch_bounds__(256)
void cp_weight_prep(const float* __restrict__ k0, const float* __restrict__ k1,
                    const float* __restrict__ k2, const float* __restrict__ k3,
                    _Float16* __restrict__ wpk)
{
    int t = blockIdx.x * 256 + threadIdx.x;
    if (t >= 9 * 32 * 32) return;
    int ic  = t & 31;
    int oc  = (t >> 5) & 31;
    int tap = t >> 10;

    int i = oc >> 4, j = (oc >> 2) & 3, kk = oc & 3;
    int a = ic >> 4, b = (ic >> 2) & 3, cc = ic & 3;

    float s = 0.f;
#pragma unroll
    for (int r = 0; r < 8; ++r) {
        s += k0[(i * 2 + a) * 8 + r] * k1[(j * 4 + b) * 8 + r] *
             k2[(kk * 4 + cc) * 8 + r] * k3[tap * 8 + r];
    }
    wpk[t] = (_Float16)s;
}

// ---------------------------------------------------------------
// Kernel 2: implicit-GEMM 3x3 conv via v_wmma_f32_16x16x32_f16
//   A (16x32 f16) = weights: M = 16 out-ch, K = 32 in-ch
//   B (32x16 f16) = activations: K = 32 in-ch, N = 16 pixels
//   C (16x16 f32): rows = out-ch, cols = pixels  -> coalesced stores
// Block: 256 thr = 8 waves; each wave owns one output row,
//        128 pixels, 32 out-ch: 8 M-tiles x 2 oc-tiles = 16 v8f accs.
// ---------------------------------------------------------------
__global__ __launch_bounds__(256)
void conv_wmma_kernel(const float* __restrict__ x,
                      const _Float16* __restrict__ wpk,
                      const float* __restrict__ bias,
                      float* __restrict__ out)
{
    __shared__ __align__(16) _Float16 smem[XS_HALVES + WL_HALVES];
    _Float16* xs = smem;              // [row 0..9][px 0..129][ch 0..31] stride-40
    _Float16* wl = smem + XS_HALVES;  // [tap 0..8][oc 0..31][ch 0..31] stride-40

    const int tid = threadIdx.x;
    const int w0  = blockIdx.x * TILE_W;   // 0..3  -> w origin
    const int h0  = blockIdx.y * TILE_H;   // 0..63 -> h origin
    const int n   = blockIdx.z;            // batch

    // ---- stage x halo tile -> LDS (f32 -> f16), coalesced along px ----
    for (int t = tid; t < HALO * 32 * PXS; t += 256) {
        int px = t % PXS;
        int rc = t / PXS;          // (row, ch) pair, px-contiguous per pair
        int c  = rc & 31;
        int r  = rc >> 5;
        int gy = h0 - 1 + r;
        int gx = w0 - 1 + px;
        float v = 0.f;
        if ((unsigned)gy < 512u && (unsigned)gx < 512u)
            v = x[(((size_t)n * 32 + c) * 512 + gy) * 512 + gx];
        xs[(r * PXS + px) * PXSTRIDE + c] = (_Float16)v;
    }
    // ---- stage packed weights -> LDS (repad to stride 40) ----
    for (int t = tid; t < 9 * 32 * 32; t += 256) {
        int ch = t & 31;
        int oc = (t >> 5) & 31;
        int tp = t >> 10;
        wl[(tp * 32 + oc) * WSTRIDE + ch] = wpk[t];
    }
    __syncthreads();

    const int wave  = tid >> 5;       // output row within tile
    const int lane  = tid & 31;
    const int lhalf = lane >> 4;      // 0 | 1  (K-half / M-half selector)
    const int l15   = lane & 15;

    v8f acc[8][2];
#pragma unroll
    for (int m = 0; m < 8; ++m)
#pragma unroll
        for (int o = 0; o < 2; ++o) {
            v8f z = {0.f, 0.f, 0.f, 0.f, 0.f, 0.f, 0.f, 0.f};
            acc[m][o] = z;
        }

#pragma unroll
    for (int tap = 0; tap < 9; ++tap) {
        const int dy = tap / 3, dx = tap % 3;
        // A fragments (ISA 16-bit A 16x32 layout):
        //   lane<16 : M=lane, K = 0..7 then 16..23
        //   lane>=16: M=lane-16, K = 8..15 then 24..31
        const _Float16* wa0 = &wl[(tap * 32 + l15) * WSTRIDE + lhalf * 8];
        const _Float16* wa1 = &wl[(tap * 32 + 16 + l15) * WSTRIDE + lhalf * 8];
        v8h a0lo = *(const v8h*)(wa0);
        v8h a0hi = *(const v8h*)(wa0 + 16);
        v8h a1lo = *(const v8h*)(wa1);
        v8h a1hi = *(const v8h*)(wa1 + 16);
        v16h a0 = __builtin_shufflevector(a0lo, a0hi, 0,1,2,3,4,5,6,7,8,9,10,11,12,13,14,15);
        v16h a1 = __builtin_shufflevector(a1lo, a1hi, 0,1,2,3,4,5,6,7,8,9,10,11,12,13,14,15);

        const int row = wave + dy;    // staged row index (gy = h0-1+row)
#pragma unroll
        for (int mt = 0; mt < 8; ++mt) {
            // B fragment (ISA 16-bit B 32x16 layout):
            //   lane<16 : N=lane (pixel), K = 0..7 then 16..23 ; lane>=16: same pixels, K = 8..15 then 24..31
            const int p = mt * 16 + l15 + dx;       // <= 129, in range
            const _Float16* xb = &xs[(row * PXS + p) * PXSTRIDE + lhalf * 8];
            v8h blo = *(const v8h*)(xb);
            v8h bhi = *(const v8h*)(xb + 16);
            v16h bf = __builtin_shufflevector(blo, bhi, 0,1,2,3,4,5,6,7,8,9,10,11,12,13,14,15);

            acc[mt][0] = __builtin_amdgcn_wmma_f32_16x16x32_f16(
                false, a0, false, bf, (short)0, acc[mt][0], false, false);
            acc[mt][1] = __builtin_amdgcn_wmma_f32_16x16x32_f16(
                false, a1, false, bf, (short)0, acc[mt][1], false, false);
        }
    }

    // ---- store: C VGPR v -> oc = oct*16 + v + (lane>=16 ? 8 : 0), col = pixel ----
    const int y = h0 + wave;
#pragma unroll
    for (int oct = 0; oct < 2; ++oct) {
        const int ocb = oct * 16 + lhalf * 8;
        float* po = out + (((size_t)n * 32 + ocb) * 512 + y) * 512 + w0 + l15;
#pragma unroll
        for (int v = 0; v < 8; ++v) {
            const float bv = bias[ocb + v];
#pragma unroll
            for (int mt = 0; mt < 8; ++mt)
                po[(size_t)v * 262144 + mt * 16] = acc[mt][oct][v] + bv;
        }
    }
}

extern "C" void kernel_launch(void* const* d_in, const int* in_sizes, int n_in,
                              void* d_out, int out_size, void* d_ws, size_t ws_size,
                              hipStream_t stream) {
    const float* x    = (const float*)d_in[0];
    const float* k0   = (const float*)d_in[1];
    const float* k1   = (const float*)d_in[2];
    const float* k2   = (const float*)d_in[3];
    const float* k3   = (const float*)d_in[4];
    const float* bias = (const float*)d_in[5];
    float*       out  = (float*)d_out;
    _Float16*    wpk  = (_Float16*)d_ws;     // 9216 f16 = 18 KB scratch

    cp_weight_prep<<<36, 256, 0, stream>>>(k0, k1, k2, k3, wpk);
    // grid: (W/128, H/8, N) = (4, 64, 8)
    conv_wmma_kernel<<<dim3(4, 64, 8), 256, 0, stream>>>(x, wpk, bias, out);
}
